// Attention_6382321402080
// MI455X (gfx1250) — matrix operand order repaired and checked by reading, not compiled, not run on hardware
//
#include <hip/hip_runtime.h>
#include <stdint.h>

// ---------------------------------------------------------------------------
// Types for CDNA5 WMMA (wave32): bf16 16x16x32, fp32 accumulate
// ---------------------------------------------------------------------------
typedef __attribute__((ext_vector_type(16))) __bf16 v16bf;
typedef __attribute__((ext_vector_type(8)))  __bf16 v8bf;
typedef __attribute__((ext_vector_type(8)))  float  v8f;
typedef int vs4i __attribute__((vector_size(16)));  // matches builtin's V4i param

#if __has_builtin(__builtin_amdgcn_global_load_async_to_lds_b128)
#define HAVE_ASYNC_LDS 1
#endif
#if __has_builtin(__builtin_amdgcn_s_wait_asynccnt)
#define HAVE_WAIT_ASYNC 1
#endif

#define BATCH 8
#define MDIM  4096
#define NDIM  4096
#define CDIM  256

// ---------------------------------------------------------------------------
// Fragment helpers
// ---------------------------------------------------------------------------
__device__ __forceinline__ v16bf cat16(v8bf lo, v8bf hi) {
  return __builtin_shufflevector(lo, hi, 0,1,2,3,4,5,6,7,8,9,10,11,12,13,14,15);
}

// A-matrix 16x32 bf16 fragment from bf16 row pointer (per-lane row).
// Lane L (L<16): row M=L, K = kk+0..15; lane L+16: K = kk+16..31.
__device__ __forceinline__ v16bf load_a_bf(const __bf16* rowp, int kk, int lane) {
  const int half = lane >> 4;
  v8bf lo = *(const v8bf*)(rowp + kk + half * 16);
  v8bf hi = *(const v8bf*)(rowp + kk + 8 + half * 16);
  return cat16(lo, hi);
}

// B-matrix 32x16 fragment from bf16 "row of K" pointer (row-major W[col][k] or
// K[key][c]): lanes 0-15 hold K=kk..kk+15 for N=lane; lanes 16-31 K=kk+16..kk+31.
__device__ __forceinline__ v16bf load_b_bf(const __bf16* wrow, int kk, int lane) {
  const __bf16* p = wrow + kk + (lane >> 4) * 16;
  return cat16(*(const v8bf*)p, *(const v8bf*)(p + 8));
}

// B-matrix fragment from LDS: per-lane 16 contiguous bf16 (caller applies half*16).
__device__ __forceinline__ v16bf load_b_lds(const __bf16* p) {
  return cat16(*(const v8bf*)p, *(const v8bf*)(p + 8));
}

// 16-byte global -> LDS copy, async via gfx1250 global_load_async_to_lds_b128.
__device__ __forceinline__ void cp16(__bf16* lds, const __bf16* g) {
#ifdef HAVE_ASYNC_LDS
  __builtin_amdgcn_global_load_async_to_lds_b128(
      (__attribute__((address_space(1))) vs4i*)(unsigned long long)(const void*)g,
      (__attribute__((address_space(3))) vs4i*)(unsigned int)(unsigned long long)(const void*)lds,
      0, 0);
#else
  *(v8bf*)lds = *(const v8bf*)g;
#endif
}

// ---------------------------------------------------------------------------
// Kernel 0: one-shot fp32 -> bf16 conversion (8 elements/thread, 16B stores)
// ---------------------------------------------------------------------------
__global__ void __launch_bounds__(256)
cvt_kernel(const float* __restrict__ src, __bf16* __restrict__ dst, long long n) {
  long long i = ((long long)blockIdx.x * blockDim.x + threadIdx.x) * 8;
  if (i + 8 <= n) {
    v8bf o;
#pragma unroll
    for (int j = 0; j < 8; ++j) o[j] = (__bf16)src[i + j];
    *(v8bf*)(dst + i) = o;
  }
}

// ---------------------------------------------------------------------------
// Kernel 1: out = in @ W^T + b  (bf16 in/W, bf16 out; optionally transposed)
// grid: (R/64, CDIM/16, BATCH), block: 128 (4 waves, one 16x16 tile each)
// ---------------------------------------------------------------------------
__global__ void __launch_bounds__(128)
proj_kernel(const __bf16* __restrict__ X, const __bf16* __restrict__ W,
            const float* __restrict__ bias, __bf16* __restrict__ Out,
            int R, int transpose_out) {
  const int lane = threadIdx.x & 31;
  const int wave = threadIdx.x >> 5;
  const int b  = blockIdx.z;
  const int r0 = (blockIdx.x * 4 + wave) * 16;
  const int c0 = blockIdx.y * 16;

  const __bf16* xrow = X + ((size_t)b * R + r0 + (lane & 15)) * CDIM;
  const __bf16* wrow = W + (size_t)(c0 + (lane & 15)) * CDIM;

  const float bv = bias[c0 + (lane & 15)];
  v8f acc;
#pragma unroll
  for (int i = 0; i < 8; ++i) acc[i] = bv;

#pragma unroll
  for (int kk = 0; kk < CDIM; kk += 32) {
    v16bf a  = load_a_bf(xrow, kk, lane);
    v16bf bm = load_b_bf(wrow, kk, lane);
    acc = __builtin_amdgcn_wmma_f32_16x16x32_bf16(false, a, false, bm,
                                                  (short)0, acc, false, false);
  }

  const int mrow = r0 + 8 * (lane >> 4);
  const int ncol = c0 + (lane & 15);
  if (!transpose_out) {
    __bf16* o = Out + (size_t)b * R * CDIM;
#pragma unroll
    for (int r = 0; r < 8; ++r) o[(size_t)(mrow + r) * CDIM + ncol] = (__bf16)acc[r];
  } else {
    __bf16* o = Out + (size_t)b * CDIM * R;  // Vt[b][c][n]
#pragma unroll
    for (int r = 0; r < 8; ++r) o[(size_t)ncol * R + (mrow + r)] = (__bf16)acc[r];
  }
}

// ---------------------------------------------------------------------------
// Kernel 2: flash attention, double-buffered async-LDS staging.
// grid: (MDIM/64, BATCH), block: 128 (4 waves). Each wave: 16 Q rows, C=256.
// ---------------------------------------------------------------------------
__device__ __forceinline__ void stage_tiles(const __bf16* Kbase, const __bf16* Vbase,
                                            int n0, __bf16* ktb, __bf16* vtb, int tid) {
  // K tile: 32 key rows x 256 cols, contiguous 16 KB in global.
  const __bf16* kg = Kbase + (size_t)n0 * CDIM;
#pragma unroll
  for (int i = 0; i < 8; ++i) {
    int e = (tid * 8 + i) * 8;  // bf16 element index, 16B chunks
    cp16(ktb + e, kg + e);
  }
  // Vt tile: 256 rows x 32 keys, 64B per row strided in global.
#pragma unroll
  for (int i = 0; i < 8; ++i) {
    int ch = tid * 8 + i;       // 1024 chunks = 256 rows x 4
    int cr = ch >> 2, co = (ch & 3) * 8;
    cp16(vtb + cr * 32 + co, Vbase + (size_t)cr * NDIM + n0 + co);
  }
}

__global__ void __launch_bounds__(128)
flash_attn_kernel(const __bf16* __restrict__ Qb, const __bf16* __restrict__ Kb,
                  const __bf16* __restrict__ Vt, __bf16* __restrict__ Hb) {
  const int lane = threadIdx.x & 31;
  const int wave = threadIdx.x >> 5;
  const int b  = blockIdx.y;
  const int m0 = blockIdx.x * 64 + wave * 16;

  __shared__ __bf16 kt[2][32 * CDIM];  // 2 x 16 KB
  __shared__ __bf16 vt[2][CDIM * 32];  // 2 x 16 KB
  __shared__ __bf16 pb[4][16 * 32];    // per-wave P staging, 4 KB

  const __bf16* Kbase = Kb + (size_t)b * NDIM * CDIM;
  const __bf16* Vbase = Vt + (size_t)b * CDIM * NDIM;

  // Q fragments resident in registers (8 x A-frag covering K=0..255)
  const __bf16* qrow = Qb + ((size_t)b * MDIM + m0 + (lane & 15)) * CDIM;
  v16bf qf[8];
#pragma unroll
  for (int kk = 0; kk < 8; ++kk) qf[kk] = load_a_bf(qrow, kk * 32, lane);

  const v8f zero8 = {};
  v8f o[16];
#pragma unroll
  for (int t = 0; t < 16; ++t) o[t] = zero8;

  float m_run[8], l_run[8];
#pragma unroll
  for (int r = 0; r < 8; ++r) { m_run[r] = -1e30f; l_run[r] = 0.f; }

  const float scale = 0.0625f;  // C^-0.5 = 1/16

  // Prologue: stage tile 0 into buffer 0 (16 async ops per wave).
  stage_tiles(Kbase, Vbase, 0, kt[0], vt[0], threadIdx.x);

  for (int n0 = 0; n0 < NDIM; n0 += 32) {
    const int cur = (n0 >> 5) & 1;

    // Issue next tile's async copies into the other buffer (its last readers
    // finished at the end-of-iteration barrier of the previous tile), then wait only
    // for the current tile's 16 in-order async ops.
    if (n0 + 32 < NDIM)
      stage_tiles(Kbase, Vbase, n0 + 32, kt[cur ^ 1], vt[cur ^ 1], threadIdx.x);
#ifdef HAVE_WAIT_ASYNC
    if (n0 + 32 < NDIM) __builtin_amdgcn_s_wait_asynccnt(16);
    else                __builtin_amdgcn_s_wait_asynccnt(0);
#endif
    __syncthreads();

    const __bf16* ktc = kt[cur];
    const __bf16* vtc = vt[cur];

    // ---- S = Q K^T (two 16x16 N-tiles, K-loop of 8 WMMAs each) ----
    v8f s[2];
    s[0] = zero8; s[1] = zero8;
#pragma unroll
    for (int kk = 0; kk < 8; ++kk) {
#pragma unroll
      for (int t = 0; t < 2; ++t) {
        const __bf16* kp = &ktc[(size_t)(t * 16 + (lane & 15)) * CDIM + kk * 32 + (lane >> 4) * 16];
        v16bf bf = load_b_lds(kp);
        s[t] = __builtin_amdgcn_wmma_f32_16x16x32_bf16(false, qf[kk], false, bf,
                                                       (short)0, s[t], false, false);
      }
    }

    // ---- online softmax (row stats live per half-wave: 4 shfl_xor steps) ----
    float mnew[8], alpha[8];
#pragma unroll
    for (int r = 0; r < 8; ++r) {
      s[0][r] *= scale; s[1][r] *= scale;
      float v = fmaxf(s[0][r], s[1][r]);
#pragma unroll
      for (int d = 1; d < 16; d <<= 1) v = fmaxf(v, __shfl_xor(v, d, 32));
      mnew[r] = fmaxf(m_run[r], v);
    }
#pragma unroll
    for (int r = 0; r < 8; ++r) {
      alpha[r] = __expf(m_run[r] - mnew[r]);
      float p0 = __expf(s[0][r] - mnew[r]);
      float p1 = __expf(s[1][r] - mnew[r]);
      s[0][r] = p0; s[1][r] = p1;
      float rs = p0 + p1;
#pragma unroll
      for (int d = 1; d < 16; d <<= 1) rs += __shfl_xor(rs, d, 32);
      l_run[r] = l_run[r] * alpha[r] + rs;
      m_run[r] = mnew[r];
    }

    // ---- C-layout -> A-layout for P via per-wave LDS staging.
    // pb is strictly per-wave and DS ops from one wave complete in order,
    // so no block barrier is needed between the stores and the reload.
    __bf16* pw = pb[wave];
#pragma unroll
    for (int t = 0; t < 2; ++t)
#pragma unroll
      for (int r = 0; r < 8; ++r)
        pw[(r + 8 * (lane >> 4)) * 32 + t * 16 + (lane & 15)] = (__bf16)s[t][r];
    v16bf pf = load_a_bf(pb[wave] + (size_t)(lane & 15) * 32, 0, lane);

    // ---- O = O*alpha + P @ V (16 WMMAs across head dim) ----
#pragma unroll
    for (int t = 0; t < 16; ++t) {
#pragma unroll
      for (int r = 0; r < 8; ++r) o[t][r] *= alpha[r];
      const __bf16* vp = &vtc[(size_t)(t * 16 + (lane & 15)) * 32 + (lane >> 4) * 16];
      v16bf vf = load_b_lds(vp);
      o[t] = __builtin_amdgcn_wmma_f32_16x16x32_bf16(false, pf, false, vf,
                                                     (short)0, o[t], false, false);
    }
    __syncthreads();  // all waves done reading kt/vt[cur] before it is reused
  }

  // ---- normalize and emit H in bf16 (row-major) ----
  __bf16* hb = Hb + (size_t)b * MDIM * CDIM;
#pragma unroll
  for (int r = 0; r < 8; ++r) {
    const float inv = 1.f / l_run[r];
    const int m = m0 + r + 8 * (lane >> 4);
#pragma unroll
    for (int t = 0; t < 16; ++t)
      hb[(size_t)m * CDIM + t * 16 + (lane & 15)] = (__bf16)(o[t][r] * inv);
  }
}

// ---------------------------------------------------------------------------
// Kernel 3: out = LayerNorm(x + H @ Wo^T + bo) * gamma + beta (fp32 out)
// grid: (MDIM/64, BATCH), block: 128. Each wave: 16 rows x full 256 cols.
// ---------------------------------------------------------------------------
__global__ void __launch_bounds__(128)
out_ln_kernel(const __bf16* __restrict__ Hb, const __bf16* __restrict__ Wob,
              const float* __restrict__ bo, const float* __restrict__ X,
              const float* __restrict__ gamma, const float* __restrict__ beta,
              float* __restrict__ Out) {
  const int lane = threadIdx.x & 31;
  const int wave = threadIdx.x >> 5;
  const int b  = blockIdx.y;
  const int m0 = blockIdx.x * 64 + wave * 16;

  const __bf16* hrow = Hb + ((size_t)b * MDIM + m0 + (lane & 15)) * CDIM;
  const __bf16* wrow = Wob + (size_t)(lane & 15) * CDIM;  // + t*16*CDIM per tile

  v8f acc[16];
#pragma unroll
  for (int t = 0; t < 16; ++t) {
    const float bv = bo[t * 16 + (lane & 15)];
#pragma unroll
    for (int i = 0; i < 8; ++i) acc[t][i] = bv;
  }

#pragma unroll
  for (int kk = 0; kk < 8; ++kk) {
    v16bf a = load_a_bf(hrow, kk * 32, lane);
#pragma unroll
    for (int t = 0; t < 16; ++t) {
      v16bf bw = load_b_bf(wrow + (size_t)t * 16 * CDIM, kk * 32, lane);
      acc[t] = __builtin_amdgcn_wmma_f32_16x16x32_bf16(false, a, false, bw,
                                                       (short)0, acc[t], false, false);
    }
  }

  const float* xb = X + (size_t)b * MDIM * CDIM;
  float* ob = Out + (size_t)b * MDIM * CDIM;
#pragma unroll
  for (int r = 0; r < 8; ++r) {
    const int m = m0 + r + 8 * (lane >> 4);
    float s = 0.f;
#pragma unroll
    for (int t = 0; t < 16; ++t) {
      float z = acc[t][r] + xb[(size_t)m * CDIM + t * 16 + (lane & 15)];
      acc[t][r] = z;
      s += z;
    }
#pragma unroll
    for (int d = 1; d < 16; d <<= 1) s += __shfl_xor(s, d, 32);
    const float mu = s * (1.f / 256.f);
    float s2 = 0.f;
#pragma unroll
    for (int t = 0; t < 16; ++t) {
      float dlt = acc[t][r] - mu;
      s2 += dlt * dlt;
    }
#pragma unroll
    for (int d = 1; d < 16; d <<= 1) s2 += __shfl_xor(s2, d, 32);
    const float rstd = rsqrtf(s2 * (1.f / 256.f) + 1e-5f);
#pragma unroll
    for (int t = 0; t < 16; ++t) {
      const int c = t * 16 + (lane & 15);
      ob[(size_t)m * CDIM + c] = (acc[t][r] - mu) * rstd * gamma[c] + beta[c];
    }
  }
}

// ---------------------------------------------------------------------------
// Host launcher
// ---------------------------------------------------------------------------
extern "C" void kernel_launch(void* const* d_in, const int* in_sizes, int n_in,
                              void* d_out, int out_size, void* d_ws, size_t ws_size,
                              hipStream_t stream) {
  const float* x     = (const float*)d_in[0];
  const float* y     = (const float*)d_in[1];
  const float* Wq    = (const float*)d_in[2];
  const float* bq    = (const float*)d_in[3];
  const float* Wk    = (const float*)d_in[4];
  const float* bk    = (const float*)d_in[5];
  const float* Wv    = (const float*)d_in[6];
  const float* bv    = (const float*)d_in[7];
  const float* Wo    = (const float*)d_in[8];
  const float* bo    = (const float*)d_in[9];
  const float* gamma = (const float*)d_in[10];
  const float* beta  = (const float*)d_in[11];
  float* out = (float*)d_out;

  char* ws = (char*)d_ws;
  const size_t seg  = (size_t)BATCH * MDIM * CDIM * sizeof(__bf16);  // 16 MB
  const size_t wseg = (size_t)CDIM * CDIM * sizeof(__bf16);          // 128 KB
  __bf16* Qb  = (__bf16*)(ws + 0 * seg);
  __bf16* Kb  = (__bf16*)(ws + 1 * seg);
  __bf16* Vt  = (__bf16*)(ws + 2 * seg);  // [B, C, N]
  __bf16* Hb  = (__bf16*)(ws + 3 * seg);
  __bf16* Xb  = (__bf16*)(ws + 4 * seg);
  __bf16* Yb  = (__bf16*)(ws + 5 * seg);
  __bf16* Wqb = (__bf16*)(ws + 6 * seg + 0 * wseg);
  __bf16* Wkb = (__bf16*)(ws + 6 * seg + 1 * wseg);
  __bf16* Wvb = (__bf16*)(ws + 6 * seg + 2 * wseg);
  __bf16* Wob = (__bf16*)(ws + 6 * seg + 3 * wseg);

  // one-shot fp32 -> bf16 conversions
  const long long nxy = (long long)BATCH * MDIM * CDIM;  // 8.4M
  const long long nw  = (long long)CDIM * CDIM;          // 65536
  cvt_kernel<<<dim3((unsigned)(nxy / 8 / 256)), dim3(256), 0, stream>>>(x,  Xb,  nxy);
  cvt_kernel<<<dim3((unsigned)(nxy / 8 / 256)), dim3(256), 0, stream>>>(y,  Yb,  nxy);
  cvt_kernel<<<dim3((unsigned)(nw  / 8 / 256)), dim3(256), 0, stream>>>(Wq, Wqb, nw);
  cvt_kernel<<<dim3((unsigned)(nw  / 8 / 256)), dim3(256), 0, stream>>>(Wk, Wkb, nw);
  cvt_kernel<<<dim3((unsigned)(nw  / 8 / 256)), dim3(256), 0, stream>>>(Wv, Wvb, nw);
  cvt_kernel<<<dim3((unsigned)(nw  / 8 / 256)), dim3(256), 0, stream>>>(Wo, Wob, nw);

  dim3 blk(128);
  dim3 pgrid(MDIM / 64, CDIM / 16, BATCH);
  proj_kernel<<<pgrid, blk, 0, stream>>>(Xb, Wqb, bq, Qb, MDIM, 0);
  proj_kernel<<<pgrid, blk, 0, stream>>>(Yb, Wkb, bk, Kb, NDIM, 0);
  proj_kernel<<<pgrid, blk, 0, stream>>>(Yb, Wvb, bv, Vt, NDIM, 1);

  dim3 fgrid(MDIM / 64, BATCH);
  flash_attn_kernel<<<fgrid, blk, 0, stream>>>(Qb, Kb, Vt, Hb);
  out_ln_kernel<<<fgrid, blk, 0, stream>>>(Hb, Wob, bo, x, gamma, beta, out);
}
